// MultiheadAttention_40321152975520
// MI455X (gfx1250) — matrix operand order repaired and checked by reading, not compiled, not run on hardware
//
#include <hip/hip_runtime.h>
#include <hip/hip_bf16.h>
#include <math.h>

typedef __attribute__((ext_vector_type(16))) _Float16 v16h;
typedef __attribute__((ext_vector_type(8)))  _Float16 v8h;
typedef __attribute__((ext_vector_type(8)))  float    v8f;
typedef __attribute__((ext_vector_type(4)))  float    v4f;
typedef __attribute__((ext_vector_type(4)))  int      v4i;

#define D_EMBED 1024
#define NHEAD   16
#define HDIM    64
#define BATCH   2
#define SEQ     2048
#define M_TOTAL (BATCH*SEQ)   /* 4096 */
#define QKV_N   (3*D_EMBED)   /* 3072 */
#define PADK    40            /* LDS row stride (halfs) for a 32-half k-slice, pads banks */

// ---------------------------------------------------------------------------
// Fragment loaders (NT GEMM: both operands stored row-major with K contiguous)
// A 16x32 f16 layout (ISA 7.12.2): lane half h owns K = {8h..8h+7, 16+8h..16+8h+7}
// B 32x16 f16 layout: lane = N, lane half h owns K = {8h..8h+7, 16+8h..16+8h+7}
// ---------------------------------------------------------------------------
__device__ __forceinline__ v16h cat8(v8h lo, v8h hi) {
  return __builtin_shufflevector(lo, hi, 0,1,2,3,4,5,6,7,8,9,10,11,12,13,14,15);
}

__device__ __forceinline__ v16h load_frag_a(const _Float16* A, int lda, int m0, int k0, int lane) {
  const int m = m0 + (lane & 15);
  const int h = lane >> 4;
  const _Float16* p = A + (long)m * lda + k0 + h * 8;
  v8h lo = *reinterpret_cast<const v8h*>(p);
  v8h hi = *reinterpret_cast<const v8h*>(p + 16);
  return cat8(lo, hi);
}

__device__ __forceinline__ v16h load_frag_b(const _Float16* B, int ldb, int n0, int k0, int lane) {
  const int n = n0 + (lane & 15);
  const int k = k0 + ((lane >> 4) << 3);
  const _Float16* p = B + (long)n * ldb + k;
  v8h lo = *reinterpret_cast<const v8h*>(p);
  v8h hi = *reinterpret_cast<const v8h*>(p + 16);
  return cat8(lo, hi);
}

#define WMMA_F16(a, b, c) \
  __builtin_amdgcn_wmma_f32_16x16x32_f16(false, (a), false, (b), (short)0, (c), false, false)

__device__ __forceinline__ v8f v8f_zero() {
  v8f z = {0.f, 0.f, 0.f, 0.f, 0.f, 0.f, 0.f, 0.f};
  return z;
}

// ---------------------------------------------------------------------------
// CDNA5 async global->LDS copy (ASYNCcnt-tracked, no VGPR payload).
// Builtin signature (from compiler diagnostic): (v4i AS1*, v4i AS3*, imm, imm)
// ---------------------------------------------------------------------------
typedef __attribute__((address_space(1))) v4i gv4i;
typedef __attribute__((address_space(3))) v4i lv4i;

__device__ __forceinline__ void async_copy_b128(const _Float16* gsrc, _Float16* ldst) {
#if defined(__has_builtin) && __has_builtin(__builtin_amdgcn_global_load_async_to_lds_b128)
  __builtin_amdgcn_global_load_async_to_lds_b128((gv4i*)gsrc, (lv4i*)ldst, 0, 0);
#else
  typedef __attribute__((address_space(3))) _Float16 lhalf;
  unsigned loff = (unsigned)(size_t)(lhalf*)ldst;
  asm volatile("global_load_async_to_lds_b128 %0, %1, off"
               :: "v"(loff), "v"(gsrc) : "memory");
#endif
}

__device__ __forceinline__ void wait_async0() {
#if defined(__has_builtin) && __has_builtin(__builtin_amdgcn_s_wait_asynccnt)
  __builtin_amdgcn_s_wait_asynccnt(0);
#else
  asm volatile("s_wait_asynccnt 0x0" ::: "memory");
#endif
}

// Stage a 128-row x 32-half k-slice into LDS (row stride PADK halfs).
// 128 threads: thread `tid` copies its own row = 4 async b128 chunks (64B).
__device__ __forceinline__ void stage_tile(const _Float16* __restrict__ G, int ld,
                                           int row0, int k, _Float16* L, int tid) {
#pragma unroll
  for (int cc = 0; cc < 4; ++cc) {
    const _Float16* g = G + (long)(row0 + tid) * ld + k + cc * 8;
    _Float16* l = L + tid * PADK + cc * 8;
    async_copy_b128(g, l);
  }
}

// ---------------------------------------------------------------------------
// Kernel 1: f32 -> f16 conversion, 8 elems/thread (b128 in, b128 out)
// ---------------------------------------------------------------------------
__global__ __launch_bounds__(256)
void cvt_f32_to_f16_kernel(const float* __restrict__ in, _Float16* __restrict__ out, int n8) {
  int i = blockIdx.x * blockDim.x + threadIdx.x;
  if (i < n8) {
    const v4f* p = reinterpret_cast<const v4f*>(in) + (long)i * 2;
    v4f a = p[0], b = p[1];
    v8h o;
#pragma unroll
    for (int j = 0; j < 4; ++j) { o[j] = (_Float16)a[j]; o[4 + j] = (_Float16)b[j]; }
    *(reinterpret_cast<v8h*>(out) + i) = o;
  }
}

// ---------------------------------------------------------------------------
// Shared GEMM core: 128x128 block tile, 4 waves of 64x64, K-loop of 32.
// A and B k-slices staged in LDS via async copies, double buffered.
// ---------------------------------------------------------------------------
__device__ __forceinline__ void gemm_core_128x128(
    const _Float16* __restrict__ A, const _Float16* __restrict__ B,
    int blockM, int blockN, int Kdim,
    _Float16* As, _Float16* Bs,   // [2][128*PADK] each
    v8f c[4][4]) {
  const int tid  = threadIdx.x;
  const int lane = tid & 31;
  const int wave = tid >> 5;
  const int wm = (wave & 1) * 64;
  const int wn = (wave >> 1) * 64;

#pragma unroll
  for (int mt = 0; mt < 4; ++mt)
#pragma unroll
    for (int nt = 0; nt < 4; ++nt) c[mt][nt] = v8f_zero();

  const int bufElems = 128 * PADK;

  // prologue: stage k=0 into buffer 0
  stage_tile(A, D_EMBED, blockM, 0, As, tid);
  stage_tile(B, D_EMBED, blockN, 0, Bs, tid);
  wait_async0();
  __syncthreads();

  int cur = 0;
  for (int k = 0; k < Kdim; k += 32) {
    // overlap: stage k+32 into the other buffer while computing this one
    if (k + 32 < Kdim) {
      stage_tile(A, D_EMBED, blockM, k + 32, As + (cur ^ 1) * bufElems, tid);
      stage_tile(B, D_EMBED, blockN, k + 32, Bs + (cur ^ 1) * bufElems, tid);
    }

    const _Float16* Ac = As + cur * bufElems;
    const _Float16* Bc = Bs + cur * bufElems;
    v16h a[4], bf[4];
#pragma unroll
    for (int mt = 0; mt < 4; ++mt) a[mt]  = load_frag_a(Ac, PADK, wm + mt * 16, 0, lane);
#pragma unroll
    for (int nt = 0; nt < 4; ++nt) bf[nt] = load_frag_b(Bc, PADK, wn + nt * 16, 0, lane);
#pragma unroll
    for (int mt = 0; mt < 4; ++mt)
#pragma unroll
      for (int nt = 0; nt < 4; ++nt) c[mt][nt] = WMMA_F16(a[mt], bf[nt], c[mt][nt]);

    wait_async0();     // this wave's async copies for k+32 are done
    __syncthreads();   // everyone done reading buf[cur] and staging buf[cur^1]
    cur ^= 1;
  }
}

// ---------------------------------------------------------------------------
// Kernel 2: QKV GEMM.  Y[m,n] = xh[m,:].Wqkv[n,:] + b[n]
// Epilogue scatters to Q[B,H,N,64] (scaled 1/8), K[B,H,N,64], Vt[B,H,64,N].
// ---------------------------------------------------------------------------
__global__ __launch_bounds__(128)
void qkv_gemm_kernel(const _Float16* __restrict__ X, const _Float16* __restrict__ W,
                     const float* __restrict__ bias,
                     _Float16* __restrict__ Q, _Float16* __restrict__ K,
                     _Float16* __restrict__ Vt) {
  __shared__ __align__(16) _Float16 As[2 * 128 * PADK];
  __shared__ __align__(16) _Float16 Bs[2 * 128 * PADK];

  const int lane = threadIdx.x & 31;
  const int wave = threadIdx.x >> 5;
  const int blockM = blockIdx.y * 128;
  const int blockN = blockIdx.x * 128;
  const int m0 = blockM + (wave & 1) * 64;
  const int n0 = blockN + (wave >> 1) * 64;

  v8f c[4][4];
  gemm_core_128x128(X, W, blockM, blockN, D_EMBED, As, Bs, c);

  const int half = lane >> 4, col = lane & 15;
#pragma unroll
  for (int mt = 0; mt < 4; ++mt)
#pragma unroll
    for (int nt = 0; nt < 4; ++nt)
#pragma unroll
      for (int r = 0; r < 8; ++r) {
        const int m = m0 + mt * 16 + half * 8 + r;
        const int n = n0 + nt * 16 + col;
        float v = c[mt][nt][r] + bias[n];
        const int which = n >> 10;        // 0=q 1=k 2=v
        const int rem = n & 1023;
        const int hh = rem >> 6, hd = rem & 63;
        const int b = m >> 11, row = m & 2047;
        const long idx = (((long)(b * NHEAD + hh)) * SEQ + row) * HDIM + hd;
        if (which == 0)      Q[idx] = (_Float16)(v * 0.125f);   // fold 1/sqrt(64)
        else if (which == 1) K[idx] = (_Float16)v;
        else {
          const long tidx = (((long)(b * NHEAD + hh)) * HDIM + hd) * SEQ + row;
          Vt[tidx] = (_Float16)v;                               // store V transposed
        }
      }
}

// ---------------------------------------------------------------------------
// Kernel 3: flash attention.  One wave per (b, h, 64-query block).
// ---------------------------------------------------------------------------
__global__ __launch_bounds__(32)
void attn_kernel(const _Float16* __restrict__ Q, const _Float16* __restrict__ K,
                 const _Float16* __restrict__ Vt, _Float16* __restrict__ O) {
  __shared__ __align__(16) float    S_lds[64][64];
  __shared__ __align__(16) _Float16 P_lds[64][64];
  __shared__ __align__(32) float    alpha_lds[64];
  __shared__ __align__(32) float    scale_lds[64];

  const int lane = threadIdx.x;
  const int half = lane >> 4, col = lane & 15;
  const int bh = blockIdx.y;
  const int b = bh >> 4, h = bh & 15;
  const int q0 = blockIdx.x * 64;

  const _Float16* Qbh = Q  + (long)bh * SEQ * HDIM;
  const _Float16* Kbh = K  + (long)bh * SEQ * HDIM;
  const _Float16* Vbh = Vt + (long)bh * HDIM * SEQ;

  v16h qf[4][2];
#pragma unroll
  for (int mt = 0; mt < 4; ++mt)
#pragma unroll
    for (int ks = 0; ks < 2; ++ks)
      qf[mt][ks] = load_frag_a(Qbh, HDIM, q0 + mt * 16, ks * 32, lane);

  v8f o[4][4];
#pragma unroll
  for (int mt = 0; mt < 4; ++mt)
#pragma unroll
    for (int ht = 0; ht < 4; ++ht) o[mt][ht] = v8f_zero();

  float mrow[2] = {-3.0e38f, -3.0e38f};
  float lrow[2] = {0.f, 0.f};

  for (int j = 0; j < SEQ / 64; ++j) {
    v8f s[4][4];
#pragma unroll
    for (int mt = 0; mt < 4; ++mt)
#pragma unroll
      for (int nt = 0; nt < 4; ++nt) s[mt][nt] = v8f_zero();

#pragma unroll
    for (int nt = 0; nt < 4; ++nt)
#pragma unroll
      for (int ks = 0; ks < 2; ++ks) {
        v16h kf = load_frag_b(Kbh, HDIM, j * 64 + nt * 16, ks * 32, lane);
#pragma unroll
        for (int mt = 0; mt < 4; ++mt) s[mt][nt] = WMMA_F16(qf[mt][ks], kf, s[mt][nt]);
      }

#pragma unroll
    for (int mt = 0; mt < 4; ++mt)
#pragma unroll
      for (int nt = 0; nt < 4; ++nt)
#pragma unroll
        for (int r = 0; r < 8; ++r)
          S_lds[mt * 16 + half * 8 + r][nt * 16 + col] = s[mt][nt][r];
    __syncthreads();

#pragma unroll
    for (int rr = 0; rr < 2; ++rr) {
      const int row = 2 * lane + rr;
      float rmax = mrow[rr];
      for (int kk = 0; kk < 64; ++kk) rmax = fmaxf(rmax, S_lds[row][kk]);
      const float al = __expf(mrow[rr] - rmax);
      float sum = 0.f;
      for (int kk = 0; kk < 64; ++kk) {
        const float p = __expf(S_lds[row][kk] - rmax);
        sum += p;
        P_lds[row][kk] = (_Float16)p;
      }
      lrow[rr] = lrow[rr] * al + sum;
      mrow[rr] = rmax;
      alpha_lds[row] = al;
    }
    __syncthreads();

#pragma unroll
    for (int mt = 0; mt < 4; ++mt) {
      v8f al = *reinterpret_cast<const v8f*>(&alpha_lds[mt * 16 + half * 8]);
#pragma unroll
      for (int ht = 0; ht < 4; ++ht)
#pragma unroll
        for (int r = 0; r < 8; ++r) o[mt][ht][r] *= al[r];
    }

    v16h pf[4][2];
#pragma unroll
    for (int mt = 0; mt < 4; ++mt)
#pragma unroll
      for (int kk = 0; kk < 2; ++kk)
        pf[mt][kk] = load_frag_a(&P_lds[0][0], 64, mt * 16, kk * 32, lane);

#pragma unroll
    for (int ht = 0; ht < 4; ++ht)
#pragma unroll
      for (int kk = 0; kk < 2; ++kk) {
        v16h vf = load_frag_b(Vbh, SEQ, ht * 16, j * 64 + kk * 32, lane);
#pragma unroll
        for (int mt = 0; mt < 4; ++mt) o[mt][ht] = WMMA_F16(pf[mt][kk], vf, o[mt][ht]);
      }
    __syncthreads();
  }

  scale_lds[2 * lane]     = 1.f / lrow[0];
  scale_lds[2 * lane + 1] = 1.f / lrow[1];
  __syncthreads();

#pragma unroll
  for (int mt = 0; mt < 4; ++mt) {
    v8f sc = *reinterpret_cast<const v8f*>(&scale_lds[mt * 16 + half * 8]);
#pragma unroll
    for (int ht = 0; ht < 4; ++ht)
#pragma unroll
      for (int r = 0; r < 8; ++r) {
        const int qrow = q0 + mt * 16 + half * 8 + r;
        const int hd = ht * 16 + col;
        O[((long)(b * SEQ + qrow)) * D_EMBED + h * HDIM + hd] =
            (_Float16)(o[mt][ht][r] * sc[r]);
      }
  }
}

// ---------------------------------------------------------------------------
// Kernel 4: output projection GEMM + bias, f32 result.
// ---------------------------------------------------------------------------
__global__ __launch_bounds__(128)
void proj_gemm_kernel(const _Float16* __restrict__ A, const _Float16* __restrict__ W,
                      const float* __restrict__ bias, float* __restrict__ out) {
  __shared__ __align__(16) _Float16 As[2 * 128 * PADK];
  __shared__ __align__(16) _Float16 Bs[2 * 128 * PADK];

  const int lane = threadIdx.x & 31;
  const int wave = threadIdx.x >> 5;
  const int blockM = blockIdx.y * 128;
  const int blockN = blockIdx.x * 128;
  const int m0 = blockM + (wave & 1) * 64;
  const int n0 = blockN + (wave >> 1) * 64;

  v8f c[4][4];
  gemm_core_128x128(A, W, blockM, blockN, D_EMBED, As, Bs, c);

  const int half = lane >> 4, col = lane & 15;
#pragma unroll
  for (int mt = 0; mt < 4; ++mt)
#pragma unroll
    for (int nt = 0; nt < 4; ++nt)
#pragma unroll
      for (int r = 0; r < 8; ++r) {
        const int m = m0 + mt * 16 + half * 8 + r;
        const int n = n0 + nt * 16 + col;
        out[(long)m * D_EMBED + n] = c[mt][nt][r] + bias[n];
      }
}

// ---------------------------------------------------------------------------
// kernel_launch
// ---------------------------------------------------------------------------
extern "C" void kernel_launch(void* const* d_in, const int* in_sizes, int n_in,
                              void* d_out, int out_size, void* d_ws, size_t ws_size,
                              hipStream_t stream) {
  const float* x      = (const float*)d_in[0];  // [2,2048,1024]
  const float* W_qkv  = (const float*)d_in[1];  // [3072,1024]
  const float* b_qkv  = (const float*)d_in[2];  // [3072]
  const float* W_proj = (const float*)d_in[3];  // [1024,1024]
  const float* b_proj = (const float*)d_in[4];  // [1024]
  float* out = (float*)d_out;                   // [2,2048,1024]

  char* w = (char*)d_ws;
  _Float16* xh   = (_Float16*)w;  w += (size_t)M_TOTAL * D_EMBED * 2;
  _Float16* wq   = (_Float16*)w;  w += (size_t)QKV_N * D_EMBED * 2;
  _Float16* wp   = (_Float16*)w;  w += (size_t)D_EMBED * D_EMBED * 2;
  _Float16* Qb   = (_Float16*)w;  w += (size_t)BATCH * NHEAD * SEQ * HDIM * 2;
  _Float16* Kb   = (_Float16*)w;  w += (size_t)BATCH * NHEAD * SEQ * HDIM * 2;
  _Float16* Vtb  = (_Float16*)w;  w += (size_t)BATCH * NHEAD * SEQ * HDIM * 2;
  _Float16* Ob   = (_Float16*)w;

  // 1) convert inputs to f16 (8 elems/thread)
  {
    int n8 = (M_TOTAL * D_EMBED) / 8;
    cvt_f32_to_f16_kernel<<<(n8 + 255) / 256, 256, 0, stream>>>(x, xh, n8);
    n8 = (QKV_N * D_EMBED) / 8;
    cvt_f32_to_f16_kernel<<<(n8 + 255) / 256, 256, 0, stream>>>(W_qkv, wq, n8);
    n8 = (D_EMBED * D_EMBED) / 8;
    cvt_f32_to_f16_kernel<<<(n8 + 255) / 256, 256, 0, stream>>>(W_proj, wp, n8);
  }

  // 2) QKV projection -> Q (scaled), K, V^T
  qkv_gemm_kernel<<<dim3(QKV_N / 128, M_TOTAL / 128), 128, 0, stream>>>(
      xh, wq, b_qkv, Qb, Kb, Vtb);

  // 3) flash attention -> O [B*N, D] f16
  attn_kernel<<<dim3(SEQ / 64, BATCH * NHEAD), 32, 0, stream>>>(Qb, Kb, Vtb, Ob);

  // 4) output projection -> f32 out
  proj_gemm_kernel<<<dim3(D_EMBED / 128, M_TOTAL / 128), 128, 0, stream>>>(
      Ob, wp, b_proj, out);
}
